// GroupedQueryAttention_32100585570588
// MI455X (gfx1250) — matrix operand order repaired and checked by reading, not compiled, not run on hardware
//
#include <hip/hip_runtime.h>

// ---------------------------------------------------------------------------
// GQA forward for MI455X (gfx1250, wave32).  All matmuls via
// V_WMMA_F32_16X16X32_BF16 (bf16 in, f32 accum).  ~155 GFLOP vs <100 MB of
// L2-resident traffic -> compute bound, so everything routes through WMMA.
// GEMM waves use 32x64 tiles: 12 global_load_b128 feed 8 wmmas per k-step.
// ---------------------------------------------------------------------------

typedef __bf16 bf16;
typedef __bf16 v16bf __attribute__((ext_vector_type(16)));
typedef float  v8f   __attribute__((ext_vector_type(8)));
typedef unsigned int v4u __attribute__((ext_vector_type(4)));

union Frag { v16bf v; v4u q[2]; };

constexpr int kB  = 2;
constexpr int kT  = 2048;
constexpr int kC  = 2048;
constexpr int kH  = 32;
constexpr int kHk = 8;
constexpr int kD  = 64;
constexpr int kKV = kHk * kD;   // 512
constexpr int kM  = kB * kT;    // 4096

#define DEV static __device__ __forceinline__

DEV v8f wmma_bf16(const Frag& a, const Frag& b, v8f c) {
    return __builtin_amdgcn_wmma_f32_16x16x32_bf16(
        false, a.v, false, b.v, (short)0, c, false, false);
}

// reductions within the 16-lane half (matches C/D row layout: lanes 0-15 hold
// rows r, lanes 16-31 hold rows r+8)
DEV float hmax16(float v) {
    v = fmaxf(v, __shfl_xor(v, 1, 32));
    v = fmaxf(v, __shfl_xor(v, 2, 32));
    v = fmaxf(v, __shfl_xor(v, 4, 32));
    v = fmaxf(v, __shfl_xor(v, 8, 32));
    return v;
}
DEV float hsum16(float v) {
    v += __shfl_xor(v, 1, 32);
    v += __shfl_xor(v, 2, 32);
    v += __shfl_xor(v, 4, 32);
    v += __shfl_xor(v, 8, 32);
    return v;
}

// ---------------------------------------------------------------------------
// vectorized f32 -> bf16 (8 elements / thread: 2x b128 in, 1x b128 out)
// ---------------------------------------------------------------------------
__global__ __launch_bounds__(256) void cvt_bf16_kernel(const float* __restrict__ in,
                                                       bf16* __restrict__ out, int n8) {
    const int i = blockIdx.x * 256 + threadIdx.x;
    if (i >= n8) return;
    const float4 f0 = ((const float4*)in)[i * 2 + 0];
    const float4 f1 = ((const float4*)in)[i * 2 + 1];
    union { bf16 h[8]; v4u q; } pk;
    pk.h[0] = (bf16)f0.x; pk.h[1] = (bf16)f0.y; pk.h[2] = (bf16)f0.z; pk.h[3] = (bf16)f0.w;
    pk.h[4] = (bf16)f1.x; pk.h[5] = (bf16)f1.y; pk.h[6] = (bf16)f1.z; pk.h[7] = (bf16)f1.w;
    ((v4u*)out)[i] = pk.q;
}

// ---------------------------------------------------------------------------
// transpose f32 [K,N] -> bf16 [N,K]   (LDS tiled, 32x32)
// ---------------------------------------------------------------------------
__global__ __launch_bounds__(256) void transpose_bf16_kernel(const float* __restrict__ in,
                                                             bf16* __restrict__ out,
                                                             int K, int N) {
    __shared__ float tile[32][33];
    const int n0 = blockIdx.x * 32, k0 = blockIdx.y * 32;
    const int tx = threadIdx.x, ty = threadIdx.y;
#pragma unroll
    for (int i = 0; i < 4; ++i)
        tile[ty + i * 8][tx] = in[(size_t)(k0 + ty + i * 8) * N + n0 + tx];
    __syncthreads();
#pragma unroll
    for (int i = 0; i < 4; ++i)
        out[(size_t)(n0 + ty + i * 8) * K + k0 + tx] = (bf16)tile[tx][ty + i * 8];
}

// ---------------------------------------------------------------------------
// GEMM: A[M,K] bf16 row-major  x  Bt[N,K] bf16 row-major (pre-transposed W)
// Each wave: 32 rows x 64 cols (head-aligned), K step 32 -> 8 wmmas / step
// fed by 12 b128 loads (2 A-frags reused over 4 B-frags and vice versa).
// MODE 0: f32 out (final Wo projection, writes d_out)
// MODE 1: RoPE + bf16 out, layout [B, H, T, 64]   (Q)
// MODE 2: RoPE + bf16 out, layout [B, Hk, T, 64]  (K)
// MODE 3: bf16 out transposed, layout [B, Hk, 64, T]  (V^T for P@V operand)
// ---------------------------------------------------------------------------
template <int MODE>
__global__ __launch_bounds__(128) void gemm_bf16_kernel(const bf16* __restrict__ A,
                                                        const bf16* __restrict__ Bt,
                                                        void* __restrict__ out,
                                                        int K, int N) {
    const int lane = threadIdx.x & 31;
    const int w    = threadIdx.x >> 5;
    const int ln   = lane & 15;
    const int hi   = lane >> 4;
    const int m0   = blockIdx.x * 32;
    const int n0   = (blockIdx.y * 4 + w) * 64;

    v8f acc[2][4] = {};

    const bf16* ap0 = A + (size_t)(m0 + ln) * K;
    const bf16* ap1 = A + (size_t)(m0 + 16 + ln) * K;

    for (int kk = 0; kk < K; kk += 32) {
        // A fragments: rows m0+ln / m0+16+ln, chunks at kk+hi*8 and kk+16+hi*8
        Frag af[2];
        af[0].q[0] = *(const v4u*)(ap0 + kk + hi * 8);
        af[0].q[1] = *(const v4u*)(ap0 + kk + 16 + hi * 8);
        af[1].q[0] = *(const v4u*)(ap1 + kk + hi * 8);
        af[1].q[1] = *(const v4u*)(ap1 + kk + 16 + hi * 8);
        if (kk + 32 < K) {
            __builtin_prefetch(ap0 + kk + 32, 0, 0);
            __builtin_prefetch(ap1 + kk + 32, 0, 0);
        }
        // B fragments: col = n0+tt*16+ln, chunks at kk+hi*8 and kk+16+hi*8
        Frag bfr[4];
#pragma unroll
        for (int tt = 0; tt < 4; ++tt) {
            const bf16* bp = Bt + (size_t)(n0 + tt * 16 + ln) * K + kk + hi * 8;
            bfr[tt].q[0] = *(const v4u*)bp;
            bfr[tt].q[1] = *(const v4u*)(bp + 16);
        }
#pragma unroll
        for (int rt = 0; rt < 2; ++rt)
#pragma unroll
            for (int tt = 0; tt < 4; ++tt)
                acc[rt][tt] = wmma_bf16(af[rt], bfr[tt], acc[rt][tt]);
    }

    if constexpr (MODE == 0) {
        float* O = (float*)out;
#pragma unroll
        for (int rt = 0; rt < 2; ++rt)
#pragma unroll
            for (int r = 0; r < 8; ++r) {
                const int row = m0 + rt * 16 + r + hi * 8;
#pragma unroll
                for (int tt = 0; tt < 4; ++tt)
                    O[(size_t)row * N + n0 + tt * 16 + ln] = acc[rt][tt][r];
            }
    } else if constexpr (MODE == 1 || MODE == 2) {
        // fused RoPE: partner column (j +/- 32) lives in acc[rt][tt+/-2],
        // same lane, same VGPR index -> pure register math
        bf16* O = (bf16*)out;
        const int heads = (MODE == 1) ? kH : kHk;
        const int hIdx  = n0 >> 6;
        const float inv0 = __powf(10000.0f, -(float)ln * (1.0f / 32.0f));
        const float inv1 = __powf(10000.0f, -(float)(16 + ln) * (1.0f / 32.0f));
#pragma unroll
        for (int rt = 0; rt < 2; ++rt)
#pragma unroll
            for (int r = 0; r < 8; ++r) {
                const int row = m0 + rt * 16 + r + hi * 8;
                const int t   = row & (kT - 1);
                const int b   = row >> 11;
                float sn0, cs0, sn1, cs1;
                __sincosf((float)t * inv0, &sn0, &cs0);
                __sincosf((float)t * inv1, &sn1, &cs1);
                const float a0 = acc[rt][0][r], a1 = acc[rt][1][r];
                const float a2 = acc[rt][2][r], a3 = acc[rt][3][r];
                const float o0 = a0 * cs0 - a2 * sn0;   // j = ln      (<32)
                const float o1 = a1 * cs1 - a3 * sn1;   // j = 16+ln   (<32)
                const float o2 = a2 * cs0 + a0 * sn0;   // j = 32+ln
                const float o3 = a3 * cs1 + a1 * sn1;   // j = 48+ln
                bf16* p = O + (((size_t)b * heads + hIdx) * kT + t) * kD + ln;
                p[0]  = (bf16)o0;
                p[16] = (bf16)o1;
                p[32] = (bf16)o2;
                p[48] = (bf16)o3;
            }
    } else {  // MODE 3: V^T  [B, Hk, 64, T]
        bf16* O = (bf16*)out;
        const int hk = n0 >> 6;
#pragma unroll
        for (int rt = 0; rt < 2; ++rt)
#pragma unroll
            for (int r = 0; r < 8; ++r) {
                const int row = m0 + rt * 16 + r + hi * 8;
                const int t   = row & (kT - 1);
                const int b   = row >> 11;
#pragma unroll
                for (int tt = 0; tt < 4; ++tt)
                    O[(((size_t)b * kHk + hk) * kD + tt * 16 + ln) * kT + t] =
                        (bf16)acc[rt][tt][r];
            }
    }
    (void)N;
}

// ---------------------------------------------------------------------------
// Flash attention (causal, GQA).  One wave per 16 query rows of one (b,h).
// Key blocks of 32: QK^T = 4 wmmas, online softmax in C/D register layout
// (row stats per (VGPR r, lane-half)), P via per-wave LDS relayout, P@V = 4
// wmmas into the running O accumulator.  Y written bf16 as [B*T, C].
// ---------------------------------------------------------------------------
__global__ __launch_bounds__(128) void attn_fwd_kernel(const bf16* __restrict__ Q,
                                                       const bf16* __restrict__ Kc,
                                                       const bf16* __restrict__ VT,
                                                       bf16* __restrict__ Y) {
    __shared__ bf16 Pl[4][16 * 32];   // per-wave 16x32 P staging tile

    const int lane = threadIdx.x & 31;
    const int w    = threadIdx.x >> 5;
    const int ln   = lane & 15;
    const int hi   = lane >> 4;

    const int wid = blockIdx.x * 4 + w;          // 8192 waves total
    const int qt  = wid & (kT / 16 - 1);         // 128 q-tiles per head
    const int h   = (wid >> 7) & (kH - 1);
    const int b   = wid >> 12;
    const int q0  = qt * 16;
    const int hk  = h >> 2;                      // 4 Q heads per KV head

    const bf16* qb = Q  + (((size_t)b * kH  + h ) * kT + q0) * kD;
    const bf16* kb = Kc + ((size_t)b * kHk + hk) * (size_t)kT * kD;
    const bf16* vb = VT + ((size_t)b * kHk + hk) * (size_t)kD * kT;

    // Q A-fragments (rows q0..q0+15, d split into two k-steps of 32)
    Frag qf[2];
#pragma unroll
    for (int s = 0; s < 2; ++s) {
        const bf16* qp = qb + (size_t)ln * kD + s * 32 + hi * 16;
        qf[s].q[0] = *(const v4u*)qp;
        qf[s].q[1] = *(const v4u*)(qp + 8);
    }

    v8f o[4] = {};
    float mArr[8], lArr[8];
#pragma unroll
    for (int r = 0; r < 8; ++r) { mArr[r] = -1.0e30f; lArr[r] = 0.0f; }

    const float scale = 0.125f;   // 1/sqrt(64)

    for (int s0 = 0; s0 <= q0 + 15; s0 += 32) {
        // ---- S = Q K^T for 16 rows x 32 keys ----
        v8f sa = {}, sb = {};
#pragma unroll
        for (int kk = 0; kk < 2; ++kk) {
            Frag k0f, k1f;
            const bf16* kp0 = kb + (size_t)(s0 + ln) * kD + kk * 32 + hi * 16;
            k0f.q[0] = *(const v4u*)kp0;
            k0f.q[1] = *(const v4u*)(kp0 + 8);
            const bf16* kp1 = kb + (size_t)(s0 + 16 + ln) * kD + kk * 32 + hi * 16;
            k1f.q[0] = *(const v4u*)kp1;
            k1f.q[1] = *(const v4u*)(kp1 + 8);
            sa = wmma_bf16(qf[kk], k0f, sa);
            sb = wmma_bf16(qf[kk], k1f, sb);
        }

        // ---- online softmax, entirely in C/D layout ----
#pragma unroll
        for (int r = 0; r < 8; ++r) {
            const int row = q0 + r + hi * 8;
            float e0 = sa[r] * scale;
            if (s0 + ln > row)      e0 = -3.0e38f;
            float e1 = sb[r] * scale;
            if (s0 + 16 + ln > row) e1 = -3.0e38f;
            const float bm = hmax16(fmaxf(e0, e1));
            const float mn = fmaxf(mArr[r], bm);
            const float al = __expf(mArr[r] - mn);
            mArr[r] = mn;
            e0 = __expf(e0 - mn);
            e1 = __expf(e1 - mn);
            lArr[r] = lArr[r] * al + hsum16(e0 + e1);
            o[0][r] *= al; o[1][r] *= al; o[2][r] *= al; o[3][r] *= al;
            Pl[w][(r + hi * 8) * 32 + ln]      = (bf16)e0;
            Pl[w][(r + hi * 8) * 32 + 16 + ln] = (bf16)e1;
        }

        asm volatile("s_wait_dscnt 0x0" ::: "memory");

        // P back out of LDS in A-operand layout
        Frag pf;
        const bf16* pp = &Pl[w][ln * 32 + hi * 16];
        pf.q[0] = *(const v4u*)pp;
        pf.q[1] = *(const v4u*)(pp + 8);

        // ---- O += P @ V  (V stored transposed: contiguous keys per d row) ----
#pragma unroll
        for (int tt = 0; tt < 4; ++tt) {
            Frag vf;
            const bf16* vp = vb + (size_t)(tt * 16 + ln) * kT + s0 + hi * 16;
            vf.q[0] = *(const v4u*)vp;
            vf.q[1] = *(const v4u*)(vp + 8);
            o[tt] = wmma_bf16(pf, vf, o[tt]);
        }
    }

    // ---- normalize and write Y[B*T, C] bf16 ----
#pragma unroll
    for (int r = 0; r < 8; ++r) {
        const int row   = q0 + r + hi * 8;
        const float inv = 1.0f / lArr[r];
        bf16* yp = Y + ((size_t)(b * kT + row)) * kC + h * kD + ln;
        yp[0]  = (bf16)(o[0][r] * inv);
        yp[16] = (bf16)(o[1][r] * inv);
        yp[32] = (bf16)(o[2][r] * inv);
        yp[48] = (bf16)(o[3][r] * inv);
    }
}

// ---------------------------------------------------------------------------
extern "C" void kernel_launch(void* const* d_in, const int* in_sizes, int n_in,
                              void* d_out, int out_size, void* d_ws, size_t ws_size,
                              hipStream_t stream) {
    (void)in_sizes; (void)n_in; (void)out_size; (void)ws_size;
    const float* x  = (const float*)d_in[0];
    const float* Wq = (const float*)d_in[1];
    const float* Wk = (const float*)d_in[2];
    const float* Wv = (const float*)d_in[3];
    const float* Wo = (const float*)d_in[4];

    char* p = (char*)d_ws;
    const size_t SZ_XB  = (size_t)kM * kC * 2;    // 16 MiB (also reused as Y)
    const size_t SZ_WQT = (size_t)kC * kC * 2;    //  8 MiB
    const size_t SZ_WKT = (size_t)kC * kKV * 2;   //  2 MiB
    const size_t SZ_QB  = (size_t)kB * kH * kT * kD * 2;   // 16 MiB
    const size_t SZ_KB  = (size_t)kB * kHk * kT * kD * 2;  //  4 MiB

    bf16* xb  = (bf16*)p;  p += SZ_XB;
    bf16* WqT = (bf16*)p;  p += SZ_WQT;
    bf16* WkT = (bf16*)p;  p += SZ_WKT;
    bf16* WvT = (bf16*)p;  p += SZ_WKT;
    bf16* WoT = (bf16*)p;  p += SZ_WQT;
    bf16* Qb  = (bf16*)p;  p += SZ_QB;
    bf16* Kb  = (bf16*)p;  p += SZ_KB;
    bf16* VTb = (bf16*)p;  p += SZ_KB;
    bf16* Yb  = xb;  // alias: x bf16 copy no longer needed once attention runs

    // 1) convert x, transpose+convert weights
    cvt_bf16_kernel<<<(kM * kC / 8 + 255) / 256, 256, 0, stream>>>(x, xb, kM * kC / 8);
    transpose_bf16_kernel<<<dim3(kC / 32, kC / 32), dim3(32, 8), 0, stream>>>(Wq, WqT, kC, kC);
    transpose_bf16_kernel<<<dim3(kKV / 32, kC / 32), dim3(32, 8), 0, stream>>>(Wk, WkT, kC, kKV);
    transpose_bf16_kernel<<<dim3(kKV / 32, kC / 32), dim3(32, 8), 0, stream>>>(Wv, WvT, kC, kKV);
    transpose_bf16_kernel<<<dim3(kC / 32, kC / 32), dim3(32, 8), 0, stream>>>(Wo, WoT, kC, kC);

    // 2) projections (RoPE fused into Q/K epilogues, V stored transposed)
    gemm_bf16_kernel<1><<<dim3(kM / 32, (kC / 64) / 4),  128, 0, stream>>>(xb, WqT, Qb,  kC, kC);
    gemm_bf16_kernel<2><<<dim3(kM / 32, (kKV / 64) / 4), 128, 0, stream>>>(xb, WkT, Kb,  kC, kKV);
    gemm_bf16_kernel<3><<<dim3(kM / 32, (kKV / 64) / 4), 128, 0, stream>>>(xb, WvT, VTb, kC, kKV);

    // 3) causal GQA flash attention -> Y (bf16, [B*T, C])
    attn_fwd_kernel<<<dim3((kB * kH * (kT / 16)) / 4), 128, 0, stream>>>(Qb, Kb, VTb, Yb);

    // 4) output projection -> f32 d_out
    gemm_bf16_kernel<0><<<dim3(kM / 32, (kC / 64) / 4), 128, 0, stream>>>(Yb, WoT, (float*)d_out, kC, kC);
}
